// DualGAT_GT_12403865551357
// MI455X (gfx1250) — matrix operand order repaired and checked by reading, not compiled, not run on hardware
//
#include <hip/hip_runtime.h>
#include <math.h>

#define NN    20000
#define EE    256000
#define HH    8
#define DINW  256
#define KHID  128
#define PREDW 32
#define NREL  16
#define DHW   32
#define LDA   264   // padded LDS row stride (halves)

typedef _Float16 v16h __attribute__((ext_vector_type(16)));
typedef _Float16 h8   __attribute__((ext_vector_type(8)));
typedef float    v8f  __attribute__((ext_vector_type(8)));

// ---------- helpers ----------
__device__ __forceinline__ unsigned fenc(float x) {
    unsigned u = __float_as_uint(x);
    return (u & 0x80000000u) ? ~u : (u | 0x80000000u);
}
__device__ __forceinline__ float fdec(unsigned u) {
    u = (u & 0x80000000u) ? (u & 0x7FFFFFFFu) : ~u;
    return __uint_as_float(u);
}

__device__ __forceinline__ v8f wmma16(v16h a, v16h b, v8f c) {
    return __builtin_amdgcn_wmma_f32_16x16x32_f16(false, a, false, b, (short)0, c, false, false);
}

// A fragment (16x32 f16, M x K) from LDS row base; contiguous k order, same as load_b:
// lane<16: K[k0..k0+15]; lane>=16: K[k0+16..k0+31]
__device__ __forceinline__ v16h load_a(const _Float16* Arow, int k0, int hi) {
    h8 lo = *(const h8*)(Arow + k0 + 16 * hi);
    h8 hv = *(const h8*)(Arow + k0 + 16 * hi + 8);
    v16h a;
#pragma unroll
    for (int i = 0; i < 8; i++) { a[i] = lo[i]; a[8 + i] = hv[i]; }
    return a;
}
// B fragment (32x16 f16, K x N), weights pre-transposed to [col][k]:
// lane<16: K[k0..k0+15]; lane>=16: K[k0+16..k0+31] -> one contiguous 32B load
__device__ __forceinline__ v16h load_b(const _Float16* Bcol, int k0, int hi) {
    return *(const v16h*)(Bcol + k0 + 16 * hi);
}

// ---------- prep / conversion ----------
__global__ void k_cvt_feats(const float* __restrict__ a, const float* __restrict__ b,
                            _Float16* __restrict__ oa, _Float16* __restrict__ ob) {
    int i = blockIdx.x * 256 + threadIdx.x;
    if (i < NN * DINW) { oa[i] = (_Float16)a[i]; ob[i] = (_Float16)b[i]; }
}

__global__ void k_cvt_w1(const float* __restrict__ W, _Float16* __restrict__ WT) {
    int i = blockIdx.x * 256 + threadIdx.x;
    if (i >= HH * DINW * KHID) return;
    int h = i / (DINW * KHID);
    int r = i - h * DINW * KHID;
    int d = r / KHID, c = r % KHID;
    WT[((size_t)(h * KHID + c)) * DINW + d] = (_Float16)W[i];
}

__global__ void k_cvt_wqkv(const float* __restrict__ Wq, const float* __restrict__ Wk,
                           const float* __restrict__ Wv, _Float16* __restrict__ WT) {
    int i = blockIdx.x * 256 + threadIdx.x;
    if (i >= 3 * DINW * DINW) return;
    int m = i / (DINW * DINW);
    int r = i - m * DINW * DINW;
    int c = r % DINW;
    const float* W = (m == 0) ? Wq : (m == 1) ? Wk : Wv;
    WT[((size_t)(m * DINW + c)) * DINW + (r / DINW)] = (_Float16)W[r];
}

// relW[r][c] = sum_p rel_emb[r][p] * We_gt[p][c]   (only 16 distinct edge features!)
__global__ void k_rel(const float* __restrict__ rel_emb, const float* __restrict__ We,
                      float* __restrict__ relW) {
    int i = blockIdx.x * 256 + threadIdx.x;
    if (i >= NREL * DINW) return;
    int r = i / DINW, c = i % DINW;
    float s = 0.f;
#pragma unroll
    for (int p = 0; p < PREDW; p++) s += rel_emb[r * PREDW + p] * We[p * DINW + c];
    relW[i] = s;
}

// relSA[l][r][h] = sum_p rel_emb[r][p] * sa_We[l][p][h]   (2*16*8 == 256)
__global__ void k_relsa(const float* __restrict__ rel_emb, const float* __restrict__ sa_We,
                        float* __restrict__ relSA) {
    int i = threadIdx.x;
    int l = i / (NREL * HH);
    int r = (i / HH) % NREL;
    int hh = i % HH;
    float s = 0.f;
#pragma unroll
    for (int p = 0; p < PREDW; p++) s += rel_emb[r * PREDW + p] * sa_We[(l * PREDW + p) * HH + hh];
    relSA[i] = s;
}

// ---------- structural channel: WMMA batched MLP with fused epilogue ----------
// h[n,w] = sum_c relu( (X @ Ws1[w])[n,c] + bs1[w,c] ) * Ws2[w,c] + bs2[w]
__global__ void k_struct_mlp(const _Float16* __restrict__ X16, const float* __restrict__ bs1,
                             const float* __restrict__ Ws2, const float* __restrict__ bs2,
                             const _Float16* __restrict__ W1T, float* __restrict__ hout) {
    __shared__ _Float16 As[16 * LDA];
    int n0 = blockIdx.x * 16;
    int tid = threadIdx.x;
    for (int c = tid; c < 16 * DINW / 8; c += 256) {
        int row = c / 32, k8 = c % 32;
        *(h8*)(As + row * LDA + k8 * 8) = *(const h8*)(X16 + (size_t)(n0 + row) * DINW + k8 * 8);
    }
    __syncthreads();
    int w = tid >> 5, lane = tid & 31, hi = lane >> 4, cl = lane & 15;
    const _Float16* Arow = As + cl * LDA;
    float hpart[8];
#pragma unroll
    for (int i = 0; i < 8; i++) hpart[i] = 0.f;
    for (int ct = 0; ct < 8; ct++) {
        int colg = ct * 16 + cl;
        const _Float16* Bcol = W1T + ((size_t)(w * KHID + colg)) * DINW;
        v8f acc = {};
#pragma unroll
        for (int ks = 0; ks < 8; ks++)
            acc = wmma16(load_a(Arow, ks * 32, hi), load_b(Bcol, ks * 32, hi), acc);
        float b1 = bs1[w * KHID + colg];
        float w2 = Ws2[w * KHID + colg];
#pragma unroll
        for (int v = 0; v < 8; v++) {
            float r = acc[v] + b1;
            hpart[v] += (r > 0.f ? r : 0.f) * w2;
        }
    }
#pragma unroll
    for (int m = 1; m < 16; m <<= 1) {
#pragma unroll
        for (int v = 0; v < 8; v++) hpart[v] += __shfl_xor(hpart[v], m, 32);
    }
    if (cl == 0) {
#pragma unroll
        for (int v = 0; v < 8; v++)
            hout[(size_t)(n0 + v + 8 * hi) * HH + w] = hpart[v] + bs2[w];
    }
}

// ---------- SA (GAT) layers ----------
__global__ void k_sa_init(const float* __restrict__ hb, const float* __restrict__ sw,
                          float* __restrict__ z, unsigned* __restrict__ smx,
                          float* __restrict__ ssm, float* __restrict__ o) {
    int i = blockIdx.x * 256 + threadIdx.x;
    if (i >= NN * HH) return;
    z[i] = hb[i] * sw[i & 7];
    smx[i] = 0u; ssm[i] = 0.f; o[i] = 0.f;
}

__global__ void k_sa_edge(const float* __restrict__ z, const float* __restrict__ rsa,
                          const float* __restrict__ al, const float* __restrict__ ar,
                          const int* __restrict__ esrc, const int* __restrict__ edst,
                          const int* __restrict__ etyp, float* __restrict__ eb,
                          unsigned* __restrict__ smx) {
    int e = blockIdx.x * 256 + threadIdx.x;
    if (e >= EE) return;
    int s = esrc[e], d = edst[e], t = etyp[e];
    const float* zs = z + (size_t)s * HH;
    const float* zd = z + (size_t)d * HH;
    const float* rr = rsa + t * HH;
#pragma unroll
    for (int h = 0; h < HH; h++) {
        float v = zs[h] * al[h] + zd[h] * ar[h] + rr[h];
        v = v > 0.f ? v : 0.2f * v;          // leaky_relu(., 0.2)
        eb[(size_t)e * HH + h] = v;
        atomicMax(smx + (size_t)d * HH + h, fenc(v));
    }
}

__global__ void k_soft_norm(float* __restrict__ eb, const unsigned* __restrict__ smx,
                            float* __restrict__ ssm, const int* __restrict__ edst) {
    int e = blockIdx.x * 256 + threadIdx.x;
    if (e >= EE) return;
    int d = edst[e];
#pragma unroll
    for (int h = 0; h < HH; h++) {
        float m = fdec(smx[(size_t)d * HH + h]);
        float ex = expf(eb[(size_t)e * HH + h] - m);
        eb[(size_t)e * HH + h] = ex;
        atomicAdd(ssm + (size_t)d * HH + h, ex);
    }
}

__global__ void k_sa_scatter(const float* __restrict__ eb, const float* __restrict__ ssm,
                             const float* __restrict__ z, const int* __restrict__ esrc,
                             const int* __restrict__ edst, float* __restrict__ o) {
    int e = blockIdx.x * 256 + threadIdx.x;
    if (e >= EE) return;
    int s = esrc[e], d = edst[e];
#pragma unroll
    for (int h = 0; h < HH; h++) {
        float attn = eb[(size_t)e * HH + h] / (ssm[(size_t)d * HH + h] + 1e-9f);
        atomicAdd(o + (size_t)d * HH + h, attn * z[(size_t)s * HH + h]);
    }
}

__global__ void k_sa_node(const float* __restrict__ z, const float* __restrict__ o,
                          float* __restrict__ hb, int doMean) {
    int n = blockIdx.x * 256 + threadIdx.x;
    if (n >= NN) return;
    float vals[HH], s = 0.f;
#pragma unroll
    for (int h = 0; h < HH; h++) {
        float x = o[(size_t)n * HH + h] + z[(size_t)n * HH + h];
        x = x > 0.f ? x : (expf(x) - 1.f);   // elu
        vals[h] = x; s += x;
    }
    if (doMean) {
        float m = s * 0.125f;
#pragma unroll
        for (int h = 0; h < HH; h++) hb[(size_t)n * HH + h] = m;
    } else {
#pragma unroll
        for (int h = 0; h < HH; h++) hb[(size_t)n * HH + h] = vals[h];
    }
}

// ---------- GT channel: q/k/v GEMM ----------
__global__ void k_qkv(const _Float16* __restrict__ X16, const _Float16* __restrict__ WT,
                      const float* __restrict__ bq, const float* __restrict__ bk,
                      const float* __restrict__ bv, float* __restrict__ qkv) {
    __shared__ _Float16 As[16 * LDA];
    int n0 = blockIdx.x * 16;
    int m = blockIdx.y;
    int tid = threadIdx.x;
    for (int c = tid; c < 16 * DINW / 8; c += 256) {
        int row = c / 32, k8 = c % 32;
        *(h8*)(As + row * LDA + k8 * 8) = *(const h8*)(X16 + (size_t)(n0 + row) * DINW + k8 * 8);
    }
    __syncthreads();
    int w = tid >> 5, lane = tid & 31, hi = lane >> 4, cl = lane & 15;
    const float* bias = (m == 0) ? bq : (m == 1) ? bk : bv;
    float* outp = qkv + (size_t)m * NN * DINW;
    const _Float16* Arow = As + cl * LDA;
    for (int t = 0; t < 2; t++) {
        int colg = (w * 2 + t) * 16 + cl;
        const _Float16* Bcol = WT + ((size_t)(m * DINW + colg)) * DINW;
        v8f acc = {};
#pragma unroll
        for (int ks = 0; ks < 8; ks++)
            acc = wmma16(load_a(Arow, ks * 32, hi), load_b(Bcol, ks * 32, hi), acc);
        float bb = bias[colg];
#pragma unroll
        for (int v = 0; v < 8; v++)
            outp[(size_t)(n0 + v + 8 * hi) * DINW + colg] = acc[v] + bb;
    }
}

__global__ void k_gt_init(float* __restrict__ agg, unsigned* __restrict__ smx,
                          float* __restrict__ ssm, float* __restrict__ cs,
                          float* __restrict__ cs2) {
    int i = blockIdx.x * 256 + threadIdx.x;
    if (i < NN * DINW) agg[i] = 0.f;
    if (i < NN * HH) { smx[i] = 0u; ssm[i] = 0.f; }
    if (i < DINW) { cs[i] = 0.f; cs2[i] = 0.f; }
}

__global__ void k_gt_edge(const float* __restrict__ q, const float* __restrict__ kk,
                          const float* __restrict__ rw, const int* __restrict__ esrc,
                          const int* __restrict__ edst, const int* __restrict__ etyp,
                          float* __restrict__ eb, unsigned* __restrict__ smx) {
    int e = blockIdx.x * 256 + threadIdx.x;
    if (e >= EE) return;
    int s = esrc[e], d = edst[e], t = etyp[e];
    const float4* qd = (const float4*)(q + (size_t)d * DINW);
    const float4* ks = (const float4*)(kk + (size_t)s * DINW);
    const float4* rr = (const float4*)(rw + (size_t)t * DINW);
#pragma unroll
    for (int h = 0; h < HH; h++) {
        float sc = 0.f;
#pragma unroll
        for (int j = 0; j < 8; j++) {
            float4 a = qd[h * 8 + j], b = ks[h * 8 + j], r = rr[h * 8 + j];
            sc += a.x * b.x * r.x + a.y * b.y * r.y + a.z * b.z * r.z + a.w * b.w * r.w;
        }
        sc *= 0.17677669529663687f;          // 1/sqrt(32)
        eb[(size_t)e * HH + h] = sc;
        atomicMax(smx + (size_t)d * HH + h, fenc(sc));
    }
}

__global__ void k_gt_scatter(const float* __restrict__ eb, const float* __restrict__ ssm,
                             const float* __restrict__ vv, const int* __restrict__ esrc,
                             const int* __restrict__ edst, float* __restrict__ agg) {
    int i = blockIdx.x * 256 + threadIdx.x;   // i = e*8 + h
    if (i >= EE * HH) return;
    int e = i >> 3, h = i & 7;
    int s = esrc[e], d = edst[e];
    float attn = eb[i] / (ssm[(size_t)d * HH + h] + 1e-9f);
    const float* vs = vv + (size_t)s * DINW + h * DHW;
    float* ag = agg + (size_t)d * DINW + h * DHW;
#pragma unroll
    for (int dd = 0; dd < DHW; dd++) atomicAdd(ag + dd, attn * vs[dd]);
}

__global__ void k_colstats(const float* __restrict__ agg, const float* __restrict__ fsem,
                           float* __restrict__ cs, float* __restrict__ cs2) {
    int t = threadIdx.x;
    int b = blockIdx.x;                       // 80 blocks x 250 rows
    float s = 0.f, s2 = 0.f;
    for (int n = b * 250; n < (b + 1) * 250; n++) {
        float x = agg[(size_t)n * DINW + t] + fsem[(size_t)n * DINW + t];
        s += x; s2 += x * x;
    }
    atomicAdd(cs + t, s);
    atomicAdd(cs2 + t, s2);
}

__global__ void k_final(const float* __restrict__ agg, const float* __restrict__ fsem,
                        const float* __restrict__ cs, const float* __restrict__ cs2,
                        const float* __restrict__ bng, const float* __restrict__ bnb,
                        const float* __restrict__ Wout, const float* __restrict__ bout,
                        const float* __restrict__ hbuf, const float* __restrict__ cent,
                        const float* __restrict__ gam, const float* __restrict__ bet,
                        float* __restrict__ out) {
    __shared__ float sred[256];
    int t = threadIdx.x;
    float mu = cs[t] * (1.f / NN);
    float var = cs2[t] * (1.f / NN) - mu * mu;
    float inv = rsqrtf(var + 1e-5f);
    float g = bng[t], bb = bnb[t], wo = Wout[t];
    for (int r = 0; r < 32; r++) {
        int n = blockIdx.x * 32 + r;
        float x = agg[(size_t)n * DINW + t] + fsem[(size_t)n * DINW + t];
        float xn = (x - mu) * inv * g + bb;
        xn = xn > 0.f ? xn : 0.f;
        sred[t] = xn * wo;
        __syncthreads();
        for (int ss = 128; ss > 0; ss >>= 1) {
            if (t < ss) sred[t] += sred[t + ss];
            __syncthreads();
        }
        if (t == 0) {
            float ls = sred[0] + bout[0];
            float c = cent[n];
            float a = 0.f;
#pragma unroll
            for (int h = 0; h < HH; h++) {
                float lg = 0.5f * hbuf[(size_t)n * HH + h] + 0.5f * ls;
                a += (c * gam[h] + bet[h]) * lg;
            }
            a *= 0.125f;
            out[n] = a > 0.f ? a : 0.01f * a;
        }
        __syncthreads();
    }
}

// ---------- launch ----------
extern "C" void kernel_launch(void* const* d_in, const int* in_sizes, int n_in,
                              void* d_out, int out_size, void* d_ws, size_t ws_size,
                              hipStream_t stream) {
    (void)in_sizes; (void)n_in; (void)out_size; (void)ws_size;
    const float* feats_struct   = (const float*)d_in[0];
    const float* feats_semantic = (const float*)d_in[1];
    const float* centrality     = (const float*)d_in[2];
    const float* Ws1  = (const float*)d_in[3];
    const float* bs1  = (const float*)d_in[4];
    const float* Ws2  = (const float*)d_in[5];
    const float* bs2  = (const float*)d_in[6];
    const float* rel_emb = (const float*)d_in[7];
    const float* sa_w  = (const float*)d_in[8];
    const float* sa_al = (const float*)d_in[9];
    const float* sa_ar = (const float*)d_in[10];
    const float* sa_We = (const float*)d_in[11];
    const float* Wq = (const float*)d_in[12];
    const float* bq = (const float*)d_in[13];
    const float* Wk = (const float*)d_in[14];
    const float* bk = (const float*)d_in[15];
    const float* Wv = (const float*)d_in[16];
    const float* bv = (const float*)d_in[17];
    const float* We_gt = (const float*)d_in[18];
    const float* bn_gamma = (const float*)d_in[19];
    const float* bn_beta  = (const float*)d_in[20];
    const float* Wout = (const float*)d_in[21];
    const float* bout = (const float*)d_in[22];
    const float* gam  = (const float*)d_in[23];
    const float* bet  = (const float*)d_in[24];
    const int* edge_types = (const int*)d_in[25];
    const int* edge_src   = (const int*)d_in[26];
    const int* edge_dst   = (const int*)d_in[27];

    char* ws = (char*)d_ws;
    _Float16* Xs16  = (_Float16*)(ws + 0);           // N*256 f16
    _Float16* Xm16  = (_Float16*)(ws + 10240000);    // N*256 f16
    _Float16* W1T   = (_Float16*)(ws + 20480000);    // 8*128*256 f16 (transposed)
    _Float16* WqkvT = (_Float16*)(ws + 21004288);    // 3*256*256 f16 (transposed)
    float* relWp    = (float*)(ws + 21397504);       // 16*256
    float* relSAp   = (float*)(ws + 21413888);       // 2*16*8
    float* hbuf     = (float*)(ws + 21414912);       // N*8
    float* zbuf     = (float*)(ws + 22054912);       // N*8
    float* outsa    = (float*)(ws + 22694912);       // N*8
    unsigned* segmx = (unsigned*)(ws + 23334912);    // N*8
    float* segsm    = (float*)(ws + 23974912);       // N*8
    float* ebuf     = (float*)(ws + 24614912);       // E*8
    float* qkv      = (float*)(ws + 32806912);       // 3*N*256
    float* agg      = (float*)(ws + 94246912);       // N*256
    float* colsum   = (float*)(ws + 114726912);      // 256
    float* colsumsq = (float*)(ws + 114727936);      // 256
    float* qb = qkv;
    float* kb = qkv + (size_t)NN * DINW;
    float* vb = qkv + 2 * (size_t)NN * DINW;

    // prep
    k_cvt_feats<<<(NN * DINW + 255) / 256, 256, 0, stream>>>(feats_struct, feats_semantic, Xs16, Xm16);
    k_cvt_w1<<<(HH * DINW * KHID + 255) / 256, 256, 0, stream>>>(Ws1, W1T);
    k_cvt_wqkv<<<(3 * DINW * DINW + 255) / 256, 256, 0, stream>>>(Wq, Wk, Wv, WqkvT);
    k_rel<<<(NREL * DINW + 255) / 256, 256, 0, stream>>>(rel_emb, We_gt, relWp);
    k_relsa<<<1, 256, 0, stream>>>(rel_emb, sa_We, relSAp);

    // structural channel
    k_struct_mlp<<<NN / 16, 256, 0, stream>>>(Xs16, bs1, Ws2, bs2, W1T, hbuf);
    for (int l = 0; l < 2; l++) {
        k_sa_init<<<(NN * HH + 255) / 256, 256, 0, stream>>>(hbuf, sa_w + l * HH, zbuf, segmx, segsm, outsa);
        k_sa_edge<<<EE / 256, 256, 0, stream>>>(zbuf, relSAp + l * NREL * HH, sa_al + l * HH,
                                                sa_ar + l * HH, edge_src, edge_dst, edge_types,
                                                ebuf, segmx);
        k_soft_norm<<<EE / 256, 256, 0, stream>>>(ebuf, segmx, segsm, edge_dst);
        k_sa_scatter<<<EE / 256, 256, 0, stream>>>(ebuf, segsm, zbuf, edge_src, edge_dst, outsa);
        k_sa_node<<<(NN + 255) / 256, 256, 0, stream>>>(zbuf, outsa, hbuf, (l == 0) ? 1 : 0);
    }

    // semantic channel (graph transformer)
    k_qkv<<<dim3(NN / 16, 3), 256, 0, stream>>>(Xm16, WqkvT, bq, bk, bv, qkv);
    k_gt_init<<<(NN * DINW + 255) / 256, 256, 0, stream>>>(agg, segmx, segsm, colsum, colsumsq);
    k_gt_edge<<<EE / 256, 256, 0, stream>>>(qb, kb, relWp, edge_src, edge_dst, edge_types, ebuf, segmx);
    k_soft_norm<<<EE / 256, 256, 0, stream>>>(ebuf, segmx, segsm, edge_dst);
    k_gt_scatter<<<(EE * HH) / 256, 256, 0, stream>>>(ebuf, segsm, vb, edge_src, edge_dst, agg);
    k_colstats<<<80, 256, 0, stream>>>(agg, feats_semantic, colsum, colsumsq);

    // batchnorm + output head + fusion + centrality scaling + final leaky_relu
    k_final<<<NN / 32, 256, 0, stream>>>(agg, feats_semantic, colsum, colsumsq, bn_gamma, bn_beta,
                                         Wout, bout, hbuf, centrality, gam, bet, (float*)d_out);
}
